// Attention_79809082295188
// MI455X (gfx1250) — compile-verified, matrix operand order checked by reading, not run on hardware
//
#include <hip/hip_runtime.h>
#include <math.h>

// Problem constants (from reference): x [4, 512, 4096]
constexpr int BB = 4;
constexpr int CC = 512;   // channels
constexpr int NN = 4096;  // positions
constexpr int KK = 64;    // C/8  (theta/phi dim)
constexpr int C2 = 256;   // C/2  (g / o_mid dim)

typedef __attribute__((ext_vector_type(16))) __bf16 v16bf;
typedef __attribute__((ext_vector_type(8)))  __bf16 v8bf;
typedef __attribute__((ext_vector_type(8)))  float  v8f;

// ---------------------------------------------------------------------------
// WMMA wrapper: D(16x16,f32) = A(16x32,bf16) * B(32x16,bf16) + C
// ---------------------------------------------------------------------------
__device__ inline v8f wmma_bf16(v16bf a, v16bf b, v8f c) {
#if defined(__gfx1250__)
  return __builtin_amdgcn_wmma_f32_16x16x32_bf16(
      /*neg_a=*/false, a, /*neg_b=*/false, b,
      /*c_mod=*/(short)0, c, /*reuse_a=*/false, /*reuse_b=*/false);
#else
  (void)a; (void)b;
  return c;
#endif
}

// Fragment K-index mapping (ISA 7.12.2, 16-bit A 16x32; B symmetric):
//   lanes 0-15 : elems 0..7 -> K = c0+0..7    ; elems 8..15 -> K = c0+16..23
//   lanes 16-31: elems 0..7 -> K = c0+8..15   ; elems 8..15 -> K = c0+24..31
// => per lane: two contiguous 8xbf16 (16B) chunks at (c0 + 8*half) and +16 elems.
__device__ inline v16bf load_frag16(const __bf16* p /* row + c0 + 8*half */) {
  v8bf lo = *(const v8bf*)p;
  v8bf hi = *(const v8bf*)(p + 16);
  v16bf r;
#pragma unroll
  for (int i = 0; i < 8; i++) { r[i] = lo[i]; r[8 + i] = hi[i]; }
  return r;
}

// ---------------------------------------------------------------------------
// Stage 0: conversions
// ---------------------------------------------------------------------------
__global__ void convert_f32_bf16(const float* __restrict__ src,
                                 __bf16* __restrict__ dst, int n) {
  for (int i = blockIdx.x * blockDim.x + threadIdx.x; i < n;
       i += gridDim.x * blockDim.x)
    dst[i] = (__bf16)src[i];
}

// xT[b][n][c] = x[b][c][n]  (bf16), so GEMM B-fragments are contiguous chunks
__global__ void transpose_convert_x(const float* __restrict__ x,
                                    __bf16* __restrict__ xT) {
  const int total = BB * NN * CC;
  for (int i = blockIdx.x * blockDim.x + threadIdx.x; i < total;
       i += gridDim.x * blockDim.x) {
    int c = i % CC;
    int n = (i / CC) % NN;
    int b = i / (CC * NN);
    xT[i] = (__bf16)x[(size_t)b * CC * NN + (size_t)c * NN + n];
  }
}

// ---------------------------------------------------------------------------
// Stage 1: projection GEMMs  out = W (Mout x 512) * x[b] (512 x 4096)
// One wave computes a 32(kout) x 64(n) tile (2 A-frags reuse each B-frag 2x).
// mode 0: store transposed  out[b][n][kout]      (theta/phi, A-frag friendly)
// mode 1: store natural     out[b][kout][n]      (g, B-frag friendly)
// ---------------------------------------------------------------------------
__global__ __launch_bounds__(32)
void proj_kernel(const __bf16* __restrict__ W, const __bf16* __restrict__ xT,
                 __bf16* __restrict__ out, int Mout, int mode) {
  const int lane = threadIdx.x & 31;
  const int half = lane >> 4;
  const int l15  = lane & 15;
  const int n_base = blockIdx.x * 64;
  const int m_base = blockIdx.y * 32;
  const int b      = blockIdx.z;

  const __bf16* wrow0 = W + (size_t)(m_base + l15) * CC;        // A rows 0..15
  const __bf16* wrow1 = W + (size_t)(m_base + 16 + l15) * CC;   // A rows 16..31
  const __bf16* xb    = xT + (size_t)b * NN * CC;

  v8f acc0[4] = {};
  v8f acc1[4] = {};
  for (int c0 = 0; c0 < CC; c0 += 32) {
    v16bf a0 = load_frag16(wrow0 + c0 + 8 * half);
    v16bf a1 = load_frag16(wrow1 + c0 + 8 * half);
#pragma unroll
    for (int j = 0; j < 4; j++) {
      const int n = n_base + j * 16 + l15;                      // B col = n
      v16bf bfrag = load_frag16(xb + (size_t)n * CC + c0 + 8 * half);
      acc0[j] = wmma_bf16(a0, bfrag, acc0[j]);
      acc1[j] = wmma_bf16(a1, bfrag, acc1[j]);
    }
  }
  // D layout: lane = col n (l15), vgpr i = row kout = base + i + 8*half
  if (mode == 0) {
#pragma unroll
    for (int j = 0; j < 4; j++) {
      const int n = n_base + j * 16 + l15;
      __bf16* row = out + (size_t)b * NN * Mout + (size_t)n * Mout;
      v8bf v0, v1;
#pragma unroll
      for (int i = 0; i < 8; i++) { v0[i] = (__bf16)acc0[j][i]; v1[i] = (__bf16)acc1[j][i]; }
      *(v8bf*)(row + m_base + 8 * half)      = v0;              // 16B stores
      *(v8bf*)(row + m_base + 16 + 8 * half) = v1;
    }
  } else {
#pragma unroll
    for (int j = 0; j < 4; j++) {
      const int n = n_base + j * 16 + l15;
#pragma unroll
      for (int i = 0; i < 8; i++) {
        const int k0 = m_base + i + 8 * half;
        out[(size_t)b * Mout * NN + (size_t)k0 * NN + n]        = (__bf16)acc0[j][i];
        out[(size_t)b * Mout * NN + (size_t)(k0 + 16) * NN + n] = (__bf16)acc1[j][i];
      }
    }
  }
}

// ---------------------------------------------------------------------------
// Stage 2: fused flash attention.
// One wave handles 32 queries (two 16-row WMMA tiles); streams all 4096 keys
// in chunks of 32. Each phi/g B-fragment feeds 2 WMMAs (query-tile reuse).
// S via WMMA -> LDS -> online softmax (lane == query row) -> P re-read from
// LDS in A-fragment layout -> 32 WMMAs/step accumulate O(32q x 256c).
// ---------------------------------------------------------------------------
__global__ __launch_bounds__(32)
void attn_kernel(const __bf16* __restrict__ thetaT,
                 const __bf16* __restrict__ phiT,
                 const __bf16* __restrict__ g,
                 __bf16* __restrict__ omidT) {
  __shared__ float  sbuf[32][32];   // scores: [query][key]
  __shared__ __bf16 pbuf[32][32];   // probabilities (bf16, A-frag source)
  __shared__ float  alpha_s[32];    // per-query rescale factor
  __shared__ float  rden_s[32];     // per-query 1/denominator

  const int lane = threadIdx.x & 31;
  const int half = lane >> 4;
  const int l15  = lane & 15;
  const int b    = blockIdx.y;
  const int n0   = blockIdx.x * 32;  // query tile base (32 queries)
  const int q    = lane;             // softmax row handled by this lane

  // Persistent A fragments: theta rows for 2 query tiles (K = 64 features)
  const __bf16* trow0 = thetaT + (size_t)b * NN * KK + (size_t)(n0 + l15) * KK;
  const __bf16* trow1 = trow0 + (size_t)16 * KK;
  v16bf a0[2], a1[2];
  a0[0] = load_frag16(trow0 + 0 + 8 * half);
  a1[0] = load_frag16(trow0 + 32 + 8 * half);
  a0[1] = load_frag16(trow1 + 0 + 8 * half);
  a1[1] = load_frag16(trow1 + 32 + 8 * half);

  v8f oA[16] = {};                   // queries n0..n0+15  x 256 channels
  v8f oB[16] = {};                   // queries n0+16..+31 x 256 channels
  float run_max = -3.0e38f, run_sum = 0.0f;

  const __bf16* phib = phiT + (size_t)b * NN * KK;
  const __bf16* gb   = g + (size_t)b * C2 * NN;

  for (int m0 = 0; m0 < NN; m0 += 32) {
    // Prefetch next key chunk (phi rows + g row segments) into cache
    if (m0 + 32 < NN) {
      __builtin_prefetch(phib + (size_t)(m0 + 32 + lane) * KK, 0, 0);
      __builtin_prefetch(gb + (size_t)(lane * 8) * NN + m0 + 32, 0, 0);
    }

    // ---- scores: 2 key subtiles x 2 query tiles (phi frags reused 2x) ----
#pragma unroll
    for (int t = 0; t < 2; t++) {
      const int m = m0 + t * 16 + l15;                 // B col = key
      const __bf16* prow = phib + (size_t)m * KK;
      v16bf b0 = load_frag16(prow + 0 + 8 * half);
      v16bf b1 = load_frag16(prow + 32 + 8 * half);
#pragma unroll
      for (int qt = 0; qt < 2; qt++) {
        v8f s = {};
        s = wmma_bf16(a0[qt], b0, s);
        s = wmma_bf16(a1[qt], b1, s);
#pragma unroll
        for (int i = 0; i < 8; i++)
          sbuf[qt * 16 + i + 8 * half][t * 16 + l15] = s[i];   // D: row=q, col=key
      }
    }
    __syncthreads();

    // ---- online softmax: lane q owns query row q ----
    float lmax = run_max;
    for (int j = 0; j < 32; j++) lmax = fmaxf(lmax, sbuf[q][j]);
    const float a_sc = __expf(run_max - lmax);
    float lsum = 0.0f;
    for (int j = 0; j < 32; j++) {
      const float e = __expf(sbuf[q][j] - lmax);
      lsum += e;
      pbuf[q][j] = (__bf16)e;
    }
    run_sum = run_sum * a_sc + lsum;
    run_max = lmax;
    alpha_s[q] = a_sc;
    __syncthreads();

    // ---- rescale running accumulators (per query row) ----
    float avA[8], avB[8];
#pragma unroll
    for (int i = 0; i < 8; i++) {
      avA[i] = alpha_s[i + 8 * half];
      avB[i] = alpha_s[16 + i + 8 * half];
    }
#pragma unroll
    for (int ct = 0; ct < 16; ct++)
#pragma unroll
      for (int i = 0; i < 8; i++) {
        oA[ct][i] *= avA[i];
        oB[ct][i] *= avB[i];
      }

    // ---- P as A fragments (2 x 16 queries x 32 keys) from LDS ----
    v16bf paA, paB;
    {
      const __bf16* prA = &pbuf[l15][0] + 8 * half;
      const __bf16* prB = &pbuf[16 + l15][0] + 8 * half;
      v8bf loA = *(const v8bf*)prA, hiA = *(const v8bf*)(prA + 16);
      v8bf loB = *(const v8bf*)prB, hiB = *(const v8bf*)(prB + 16);
#pragma unroll
      for (int i = 0; i < 8; i++) {
        paA[i] = loA[i]; paA[8 + i] = hiA[i];
        paB[i] = loB[i]; paB[8 + i] = hiB[i];
      }
    }

    // ---- O += P * g^T : 16 channel tiles; each g frag feeds 2 WMMAs ----
#pragma unroll
    for (int ct = 0; ct < 16; ct++) {
      const int c = ct * 16 + l15;                     // B col = channel
      v16bf gf = load_frag16(gb + (size_t)c * NN + m0 + 8 * half);
      oA[ct] = wmma_bf16(paA, gf, oA[ct]);
      oB[ct] = wmma_bf16(paB, gf, oB[ct]);
    }
    __syncthreads();   // sbuf/pbuf safe to overwrite next iteration
  }

  // ---- finalize: divide by softmax denominator, store o_midT[b][n][c] ----
  rden_s[q] = 1.0f / run_sum;
  __syncthreads();
  float dvA[8], dvB[8];
#pragma unroll
  for (int i = 0; i < 8; i++) {
    dvA[i] = rden_s[i + 8 * half];
    dvB[i] = rden_s[16 + i + 8 * half];
  }
  __bf16* ob = omidT + (size_t)b * NN * C2;
#pragma unroll
  for (int ct = 0; ct < 16; ct++) {
    const int c = ct * 16 + l15;
#pragma unroll
    for (int i = 0; i < 8; i++) {
      const int nA = n0 + i + 8 * half;
      const int nB = n0 + 16 + i + 8 * half;
      ob[(size_t)nA * C2 + c] = (__bf16)(oA[ct][i] * dvA[i]);
      ob[(size_t)nB * C2 + c] = (__bf16)(oB[ct][i] * dvB[i]);
    }
  }
}

// ---------------------------------------------------------------------------
// Stage 3: out[b][co][n] = gamma * (W_o * o_mid)[co][n] + x[b][co][n]
// 32(co) x 64(n) tile per wave; each o_mid B-frag feeds 2 WMMAs.
// ---------------------------------------------------------------------------
__global__ __launch_bounds__(32)
void outproj_kernel(const __bf16* __restrict__ Wo,
                    const __bf16* __restrict__ omidT,
                    const float* __restrict__ x,
                    const float* __restrict__ gamma_p,
                    float* __restrict__ out) {
  const int lane = threadIdx.x & 31;
  const int half = lane >> 4;
  const int l15  = lane & 15;
  const int n_base = blockIdx.x * 64;
  const int m_base = blockIdx.y * 32;
  const int b      = blockIdx.z;
  const float gamma = gamma_p[0];

  const __bf16* wrow0 = Wo + (size_t)(m_base + l15) * C2;
  const __bf16* wrow1 = Wo + (size_t)(m_base + 16 + l15) * C2;
  const __bf16* ob    = omidT + (size_t)b * NN * C2;

  v8f acc0[4] = {};
  v8f acc1[4] = {};
  for (int c0 = 0; c0 < C2; c0 += 32) {
    v16bf a0 = load_frag16(wrow0 + c0 + 8 * half);
    v16bf a1 = load_frag16(wrow1 + c0 + 8 * half);
#pragma unroll
    for (int j = 0; j < 4; j++) {
      const int n = n_base + j * 16 + l15;
      v16bf bf = load_frag16(ob + (size_t)n * C2 + c0 + 8 * half);
      acc0[j] = wmma_bf16(a0, bf, acc0[j]);
      acc1[j] = wmma_bf16(a1, bf, acc1[j]);
    }
  }
  const float* xb = x + (size_t)b * CC * NN;
  float* outb     = out + (size_t)b * CC * NN;
#pragma unroll
  for (int j = 0; j < 4; j++) {
    const int n = n_base + j * 16 + l15;
#pragma unroll
    for (int i = 0; i < 8; i++) {
      const int co = m_base + i + 8 * half;
      const size_t i0 = (size_t)co * NN + n;
      const size_t i1 = (size_t)(co + 16) * NN + n;
      outb[i0] = gamma * acc0[j][i] + xb[i0];
      outb[i1] = gamma * acc1[j][i] + xb[i1];
    }
  }
}

// ---------------------------------------------------------------------------
// Host launcher
// ---------------------------------------------------------------------------
extern "C" void kernel_launch(void* const* d_in, const int* in_sizes, int n_in,
                              void* d_out, int out_size, void* d_ws,
                              size_t ws_size, hipStream_t stream) {
  (void)in_sizes; (void)n_in; (void)out_size; (void)ws_size;
  const float* x     = (const float*)d_in[0];
  const float* Wth   = (const float*)d_in[1];
  const float* Wph   = (const float*)d_in[2];
  const float* Wg    = (const float*)d_in[3];
  const float* Wo    = (const float*)d_in[4];
  const float* gamma = (const float*)d_in[5];
  float* out = (float*)d_out;

  char* ws = (char*)d_ws;
  size_t off = 0;
  auto alloc = [&](size_t bytes) -> void* {
    void* p = ws + off;
    off += (bytes + 255) & ~(size_t)255;
    return p;
  };
  __bf16* xT     = (__bf16*)alloc((size_t)BB * NN * CC * 2);   // 16.8 MB
  __bf16* wth_bf = (__bf16*)alloc((size_t)KK * CC * 2);
  __bf16* wph_bf = (__bf16*)alloc((size_t)KK * CC * 2);
  __bf16* wg_bf  = (__bf16*)alloc((size_t)C2 * CC * 2);
  __bf16* wo_bf  = (__bf16*)alloc((size_t)CC * C2 * 2);
  __bf16* thetaT = (__bf16*)alloc((size_t)BB * NN * KK * 2);   // 2.1 MB
  __bf16* phiT   = (__bf16*)alloc((size_t)BB * NN * KK * 2);   // 2.1 MB
  __bf16* gmat   = (__bf16*)alloc((size_t)BB * C2 * NN * 2);   // 8.4 MB
  __bf16* omidT  = (__bf16*)alloc((size_t)BB * NN * C2 * 2);   // 8.4 MB

  // Stage 0: convert + transpose
  transpose_convert_x<<<8192, 256, 0, stream>>>(x, xT);
  convert_f32_bf16<<<128, 256, 0, stream>>>(Wth, wth_bf, KK * CC);
  convert_f32_bf16<<<128, 256, 0, stream>>>(Wph, wph_bf, KK * CC);
  convert_f32_bf16<<<512, 256, 0, stream>>>(Wg, wg_bf, C2 * CC);
  convert_f32_bf16<<<512, 256, 0, stream>>>(Wo, wo_bf, CC * C2);

  // Stage 1: projections (theta/phi transposed layout, g natural layout)
  proj_kernel<<<dim3(NN / 64, KK / 32, BB), 32, 0, stream>>>(wth_bf, xT, thetaT, KK, 0);
  proj_kernel<<<dim3(NN / 64, KK / 32, BB), 32, 0, stream>>>(wph_bf, xT, phiT, KK, 0);
  proj_kernel<<<dim3(NN / 64, C2 / 32, BB), 32, 0, stream>>>(wg_bf, xT, gmat, C2, 1);

  // Stage 2: fused flash attention (32 queries per wave)
  attn_kernel<<<dim3(NN / 32, BB), 32, 0, stream>>>(thetaT, phiT, gmat, omidT);

  // Stage 3: output projection + residual
  outproj_kernel<<<dim3(NN / 64, CC / 32, BB), 32, 0, stream>>>(wo_bf, omidT, x, gamma, out);
}
